// Encoding2Tokens_7206955122693
// MI455X (gfx1250) — matrix operand order repaired and checked by reading, not compiled, not run on hardware
//
#include <hip/hip_runtime.h>

// ---------------------------------------------------------------------------
// Encoding2Tokens forward for MI455X (gfx1250): all GEMMs via
// v_wmma_f32_16x16x32_bf16 (bf16 A/B, f32 accumulate). Cross-attention K/V of
// the encoder memory is computed once per layer (numerically identical to the
// reference, which redundantly recomputes it each decode step).
// Workspace requirement: ~850 MB in d_ws (bump-allocated below).
// ---------------------------------------------------------------------------

typedef unsigned short u16;
typedef __attribute__((ext_vector_type(8)))  __bf16 bf16x8;
typedef __attribute__((ext_vector_type(16))) __bf16 v16bf;
typedef __attribute__((ext_vector_type(8)))  float  v8f;

constexpr int D    = 256;    // model dim
constexpr int NH   = 2;      // heads
constexpr int HDIM = 128;    // head dim
constexpr int DFF  = 2048;   // ffn dim
constexpr int NB   = 20;     // objects
constexpr int BATCH= 32;     // batch
constexpr int SEQ  = 1024;   // encoder sequence
constexpr int DCAP = 32;     // decoder token row capacity (>= NB+1, mult of 32)
constexpr float RSQRT_HD = 0.08838834764831845f;  // 1/sqrt(128)

// ------------------------------- device helpers ----------------------------

__device__ __forceinline__ u16 f32_bf16(float f) {
  unsigned u = __float_as_uint(f);
  u += 0x7FFFu + ((u >> 16) & 1u);           // round-to-nearest-even
  return (u16)(u >> 16);
}

// A fragment, 16-bit 16x32 layout: lane L holds row (L%16); low 8 bf16 at
// K = 8*(L/16), high 8 bf16 at K = 16 + 8*(L/16)  (two 16B loads).
__device__ __forceinline__ v16bf load_a_frag(const u16* p) {
  union { v16bf v; bf16x8 h[2]; } u;
  u.h[0] = *(const bf16x8*)(p);
  u.h[1] = *(const bf16x8*)(p + 16);
  return u.v;
}

// ------------------------------- GEMM kernel -------------------------------
// C[z] = alpha * A[z] (MxK, row-major, bf16) * B[z]^T (NxK, row-major, bf16)
//        (+ bias[n]) (ReLU?)  ->  f32 or bf16, optionally stored transposed.
// Batch z decomposed as (zb, zh) = (z / Hdiv, z % Hdiv) with separate element
// offsets per tensor, covering (batch, head)-sliced attention operands.
// Block: 256 threads = 8 waves; wave tile 32x32 (2x2 WMMA fragments);
// block tile 128(M) x 64(N).

template<bool BIAS, bool RELU, bool OBF, bool TR>
__global__ __launch_bounds__(256) void e2t_gemm(
    const u16* __restrict__ A, const u16* __restrict__ Bm,
    const float* __restrict__ bias, void* __restrict__ Cv,
    int M, int N, int K, int lda, int ldb, int ldc,
    long oAb, long oAh, long oBb, long oBh, long oCb, long oCh,
    int Hdiv, float alpha)
{
  const int z  = blockIdx.z;
  const int zb = z / Hdiv, zh = z % Hdiv;
  A  += (long)zb * oAb + (long)zh * oAh;
  Bm += (long)zb * oBb + (long)zh * oBh;
  const long coff = (long)zb * oCb + (long)zh * oCh;

  const int lane = threadIdx.x & 31;
  const int wave = threadIdx.x >> 5;
  const int m0 = blockIdx.y * 128 + (wave & 3) * 32;
  const int n0 = blockIdx.x * 64  + (wave >> 2) * 32;
  const int r16  = lane & 15;
  const int half = lane >> 4;

  // clamp out-of-range rows (stores are guarded; garbage accum is discarded)
  int ma0 = m0 + r16;      if (ma0 >= M) ma0 = M - 1;
  int ma1 = m0 + 16 + r16; if (ma1 >= M) ma1 = M - 1;
  int nb0 = n0 + r16;      if (nb0 >= N) nb0 = N - 1;
  int nb1 = n0 + 16 + r16; if (nb1 >= N) nb1 = N - 1;
  const u16* pa0 = A  + (long)ma0 * lda + half * 8;
  const u16* pa1 = A  + (long)ma1 * lda + half * 8;
  // B fragment: lane holds column n, same K split as A (8*(L/16), 16+8*(L/16))
  const u16* pb0 = Bm + (long)nb0 * ldb + half * 8;
  const u16* pb1 = Bm + (long)nb1 * ldb + half * 8;

  v8f acc[2][2] = {};
  for (int k = 0; k < K; k += 32) {
    if (k + 32 < K) {       // WGP-scope prefetch of next K block (fills near $)
      __builtin_prefetch(pa0 + k + 32, 0, 3);
      __builtin_prefetch(pb0 + k + 32, 0, 3);
    }
    v16bf a0 = load_a_frag(pa0 + k);
    v16bf a1 = load_a_frag(pa1 + k);
    v16bf b0 = load_a_frag(pb0 + k);
    v16bf b1 = load_a_frag(pb1 + k);
    acc[0][0] = __builtin_amdgcn_wmma_f32_16x16x32_bf16(false, a0, false, b0,
                    (short)0, acc[0][0], false, false);
    acc[0][1] = __builtin_amdgcn_wmma_f32_16x16x32_bf16(false, a0, false, b1,
                    (short)0, acc[0][1], false, false);
    acc[1][0] = __builtin_amdgcn_wmma_f32_16x16x32_bf16(false, a1, false, b0,
                    (short)0, acc[1][0], false, false);
    acc[1][1] = __builtin_amdgcn_wmma_f32_16x16x32_bf16(false, a1, false, b1,
                    (short)0, acc[1][1], false, false);
  }

  // C/D layout: lane L -> n = L%16 ; VGPR r -> m = r + 8*(L/16)
  #pragma unroll
  for (int i = 0; i < 2; ++i) {
    #pragma unroll
    for (int j = 0; j < 2; ++j) {
      int n = n0 + j * 16 + r16;
      if (n >= N) continue;
      float bv = BIAS ? bias[n] : 0.0f;
      #pragma unroll
      for (int r = 0; r < 8; ++r) {
        int m = m0 + i * 16 + half * 8 + r;
        if (m >= M) continue;
        float v = acc[i][j][r] * alpha + bv;
        if (RELU) v = fmaxf(v, 0.0f);
        long idx = TR ? ((long)n * ldc + m) : ((long)m * ldc + n);
        if (OBF) ((u16*)Cv)[coff + idx] = f32_bf16(v);
        else     ((float*)Cv)[coff + idx] = v;
      }
    }
  }
}

// ------------------------------ softmax kernel -----------------------------
// One wave per row. Valid length = Lk (or row+1 if causal). Writes bf16 probs
// zero-padded to Lpad so the following AV GEMM has K % 32 == 0.

__global__ __launch_bounds__(256) void e2t_softmax(
    const float* __restrict__ scores, u16* __restrict__ probs,
    int rows, int Lk, int Lpad, long soff_z, long poff_z, int causal)
{
  const int z = blockIdx.z;
  const int wave = threadIdx.x >> 5, lane = threadIdx.x & 31;
  const int row = blockIdx.x * 8 + wave;
  if (row >= rows) return;
  const float* sr = scores + (long)z * soff_z + (long)row * Lpad;
  u16* pr = probs + (long)z * poff_z + (long)row * Lpad;
  const int L = causal ? (row + 1) : Lk;

  float m = -3.0e38f;
  for (int j = lane; j < L; j += 32) m = fmaxf(m, sr[j]);
  for (int o = 16; o > 0; o >>= 1) m = fmaxf(m, __shfl_xor(m, o, 32));
  float s = 0.0f;
  for (int j = lane; j < L; j += 32) s += __expf(sr[j] - m);
  for (int o = 16; o > 0; o >>= 1) s += __shfl_xor(s, o, 32);
  const float inv = 1.0f / s;
  for (int j = lane; j < Lpad; j += 32) {
    float p = (j < L) ? __expf(sr[j] - m) * inv : 0.0f;
    pr[j] = f32_bf16(p);
  }
}

// ------------------------- residual + LayerNorm kernel ---------------------
// y = LN(x + r) * w + b over D=256; one wave per row (8 floats/lane).
// Emits both f32 (next residual) and bf16 (next GEMM operand).

__global__ __launch_bounds__(256) void e2t_add_ln(
    const float* __restrict__ x, const float* __restrict__ r,
    const float* __restrict__ w, const float* __restrict__ b,
    float* __restrict__ outf, u16* __restrict__ outb,
    int rows_per_batch, long xstr, long rstr, long ostr, int total_rows)
{
  const int wave = threadIdx.x >> 5, lane = threadIdx.x & 31;
  const int row = blockIdx.x * 8 + wave;
  if (row >= total_rows) return;
  const int bb = row / rows_per_batch, t = row % rows_per_batch;
  const float* xr = x + (long)bb * xstr + (long)t * D;
  const float* rr = r + (long)bb * rstr + (long)t * D;

  float v[8]; float sum = 0.0f;
  #pragma unroll
  for (int i = 0; i < 8; ++i) { v[i] = xr[lane * 8 + i] + rr[lane * 8 + i]; sum += v[i]; }
  for (int o = 16; o > 0; o >>= 1) sum += __shfl_xor(sum, o, 32);
  const float mean = sum * (1.0f / 256.0f);
  float var = 0.0f;
  #pragma unroll
  for (int i = 0; i < 8; ++i) { float d = v[i] - mean; var += d * d; }
  for (int o = 16; o > 0; o >>= 1) var += __shfl_xor(var, o, 32);
  const float inv = rsqrtf(var * (1.0f / 256.0f) + 1e-5f);

  float* of = outf + (long)bb * ostr + (long)t * D;
  u16*   ob = outb + (long)bb * ostr + (long)t * D;
  #pragma unroll
  for (int i = 0; i < 8; ++i) {
    const int c = lane * 8 + i;
    float y = (v[i] - mean) * inv * w[c] + b[c];
    of[c] = y; ob[c] = f32_bf16(y);
  }
}

// --------------------------- small utility kernels -------------------------

__global__ void e2t_cvt(const float* __restrict__ in, u16* __restrict__ out, long n) {
  long i = (long)blockIdx.x * 256 + threadIdx.x;
  const long st = (long)gridDim.x * 256;
  for (; i < n; i += st) out[i] = f32_bf16(in[i]);
}

__global__ void e2t_copy_cvt(const float* __restrict__ in, float* __restrict__ outf,
                             u16* __restrict__ outb, long n) {
  long i = (long)blockIdx.x * 256 + threadIdx.x;
  const long st = (long)gridDim.x * 256;
  for (; i < n; i += st) { float v = in[i]; outf[i] = v; outb[i] = f32_bf16(v); }
}

__global__ void e2t_tok_init(const float* __restrict__ init, float* tokf, u16* tokb) {
  const int i = blockIdx.x * 256 + threadIdx.x;
  if (i >= BATCH * D) return;
  const int bb = i / D, c = i % D;
  const float v = init[c];
  tokf[(long)bb * DCAP * D + c] = v;
  tokb[(long)bb * DCAP * D + c] = f32_bf16(v);
}

__global__ void e2t_append(const float* __restrict__ y, int srcrow, int dstrow,
                           float* tokf, u16* tokb) {
  const int i = blockIdx.x * 256 + threadIdx.x;
  if (i >= BATCH * D) return;
  const int bb = i / D, c = i % D;
  const float v = y[(long)bb * DCAP * D + (long)srcrow * D + c];
  tokf[(long)bb * DCAP * D + (long)dstrow * D + c] = v;
  tokb[(long)bb * DCAP * D + (long)dstrow * D + c] = f32_bf16(v);
}

__global__ void e2t_emit(const float* __restrict__ tokf, float* __restrict__ out) {
  const int i = blockIdx.x * 256 + threadIdx.x;
  if (i >= BATCH * (NB + 1) * D) return;
  const int bb = i / ((NB + 1) * D), rest = i % ((NB + 1) * D);
  out[i] = tokf[(long)bb * DCAP * D + rest];
}

// ------------------------------- host helpers ------------------------------

template<bool BIAS, bool RELU, bool OBF, bool TR>
static inline void launch_gemm(hipStream_t st, const u16* A, const u16* Bm,
    const float* bias, void* C, int M, int N, int K, int lda, int ldb, int ldc,
    long oAb, long oAh, long oBb, long oBh, long oCb, long oCh,
    int Hdiv, int nz, float alpha)
{
  dim3 grid((N + 63) / 64, (M + 127) / 128, nz);
  e2t_gemm<BIAS, RELU, OBF, TR><<<grid, 256, 0, st>>>(A, Bm, bias, C,
      M, N, K, lda, ldb, ldc, oAb, oAh, oBb, oBh, oCb, oCh, Hdiv, alpha);
}

struct AttnP  { const float *in_w, *in_b, *out_w, *out_b; };
struct LayerP {
  AttnP sa, ca;
  const float *lin1_w, *lin1_b, *lin2_w, *lin2_b;
  const float *ln1_w, *ln1_b, *ln2_w, *ln2_b, *ln3_w, *ln3_b;
};
struct AttnW  { u16 *wqkv, *wo; };       // bf16; wq=wqkv, wk=+D*D, wv=+2*D*D
struct LayerW { AttnW sa, ca; u16 *w1, *w2; };

static inline void* bump(char*& p, size_t bytes) {
  void* r = (void*)p;
  p += (bytes + 255) & ~(size_t)255;
  return r;
}

struct DecScratch {
  u16 *dq, *dk, *dvT, *dattn, *dffn, *dprob, *cprob;
  float *dscore, *cscore, *dmha;
  float *x2f, *x3f; u16 *x2b, *x3b;
};

// one post-norm decoder layer (self-attn [+causal], cached cross-attn, ffn)
static void dec_layer(hipStream_t st, const LayerP& P, const LayerW& W,
                      const u16* kmem, const u16* vmem, int T, int causal,
                      const float* xinf, const u16* xinb,
                      float* outf, u16* outb, const DecScratch& s)
{
  const long SD = (long)DCAP * D;
  // ---- self attention ----
  launch_gemm<true,false,true,false>(st, xinb, W.sa.wqkv,           P.sa.in_b,       s.dq,
      T, D, D, D, D, D, SD,0, 0,0, SD,0, 1, BATCH, 1.f);
  launch_gemm<true,false,true,false>(st, xinb, W.sa.wqkv + D*D,     P.sa.in_b + D,   s.dk,
      T, D, D, D, D, D, SD,0, 0,0, SD,0, 1, BATCH, 1.f);
  launch_gemm<true,false,true,true >(st, xinb, W.sa.wqkv + 2*D*D,   P.sa.in_b + 2*D, s.dvT,
      T, D, D, D, D, BATCH*DCAP, SD,0, 0,0, (long)DCAP,0, 1, BATCH, 1.f);
  launch_gemm<false,false,false,false>(st, s.dq, s.dk, nullptr, s.dscore,
      T, T, HDIM, D, D, DCAP, SD,(long)HDIM, SD,(long)HDIM,
      (long)NH*DCAP*DCAP,(long)DCAP*DCAP, NH, BATCH*NH, RSQRT_HD);
  e2t_softmax<<<dim3((T+7)/8,1,BATCH*NH),256,0,st>>>(s.dscore, s.dprob,
      T, T, DCAP, (long)DCAP*DCAP, (long)DCAP*DCAP, causal);
  launch_gemm<false,false,true,false>(st, s.dprob, s.dvT, nullptr, s.dattn,
      T, HDIM, DCAP, DCAP, BATCH*DCAP, D,
      (long)NH*DCAP*DCAP,(long)DCAP*DCAP, (long)DCAP,(long)HDIM*BATCH*DCAP,
      SD,(long)HDIM, NH, BATCH*NH, 1.f);
  launch_gemm<true,false,false,false>(st, s.dattn, W.sa.wo, P.sa.out_b, s.dmha,
      T, D, D, D, D, D, SD,0, 0,0, SD,0, 1, BATCH, 1.f);
  e2t_add_ln<<<dim3((BATCH*T+7)/8),256,0,st>>>(xinf, s.dmha, P.ln1_w, P.ln1_b,
      s.x2f, s.x2b, T, SD, SD, SD, BATCH*T);
  // ---- cross attention (K/V of mem cached) ----
  launch_gemm<true,false,true,false>(st, s.x2b, W.ca.wqkv, P.ca.in_b, s.dq,
      T, D, D, D, D, D, SD,0, 0,0, SD,0, 1, BATCH, 1.f);
  launch_gemm<false,false,false,false>(st, s.dq, kmem, nullptr, s.cscore,
      T, SEQ, HDIM, D, D, SEQ, SD,(long)HDIM, (long)SEQ*D,(long)HDIM,
      (long)NH*DCAP*SEQ,(long)DCAP*SEQ, NH, BATCH*NH, RSQRT_HD);
  e2t_softmax<<<dim3((T+7)/8,1,BATCH*NH),256,0,st>>>(s.cscore, s.cprob,
      T, SEQ, SEQ, (long)DCAP*SEQ, (long)DCAP*SEQ, 0);
  launch_gemm<false,false,true,false>(st, s.cprob, vmem, nullptr, s.dattn,
      T, HDIM, SEQ, SEQ, BATCH*SEQ, D,
      (long)NH*DCAP*SEQ,(long)DCAP*SEQ, (long)SEQ,(long)HDIM*BATCH*SEQ,
      SD,(long)HDIM, NH, BATCH*NH, 1.f);
  launch_gemm<true,false,false,false>(st, s.dattn, W.ca.wo, P.ca.out_b, s.dmha,
      T, D, D, D, D, D, SD,0, 0,0, SD,0, 1, BATCH, 1.f);
  e2t_add_ln<<<dim3((BATCH*T+7)/8),256,0,st>>>(s.x2f, s.dmha, P.ln2_w, P.ln2_b,
      s.x3f, s.x3b, T, SD, SD, SD, BATCH*T);
  // ---- ffn ----
  launch_gemm<true,true,true,false>(st, s.x3b, W.w1, P.lin1_b, s.dffn,
      T, DFF, D, D, D, DFF, SD,0, 0,0, (long)DCAP*DFF,0, 1, BATCH, 1.f);
  launch_gemm<true,false,false,false>(st, s.dffn, W.w2, P.lin2_b, s.dmha,
      T, D, DFF, DFF, DFF, D, (long)DCAP*DFF,0, 0,0, SD,0, 1, BATCH, 1.f);
  e2t_add_ln<<<dim3((BATCH*T+7)/8),256,0,st>>>(s.x3f, s.dmha, P.ln3_w, P.ln3_b,
      outf, outb, T, SD, SD, SD, BATCH*T);
}

// --------------------------------- driver ----------------------------------

extern "C" void kernel_launch(void* const* d_in, const int* in_sizes, int n_in,
                              void* d_out, int out_size, void* d_ws, size_t ws_size,
                              hipStream_t stream) {
  (void)in_sizes; (void)n_in; (void)out_size; (void)ws_size;

  // ---- parse inputs (setup_inputs() insertion order, nested dicts in
  //      insertion order: sa{in_w,in_b,out_w,out_b}, lin*, ln1/2, [ca, ln3]) --
  int cur = 0;
  auto nxt = [&]() { return (const float*)d_in[cur++]; };
  const float* source = nxt();
  LayerP enc[2], sdec[2], dcr[2];
  auto rdA = [&](AttnP& a) { a.in_w = nxt(); a.in_b = nxt(); a.out_w = nxt(); a.out_b = nxt(); };
  auto rdL = [&](LayerP& L, bool cross) {
    rdA(L.sa);
    L.lin1_w = nxt(); L.lin1_b = nxt(); L.lin2_w = nxt(); L.lin2_b = nxt();
    L.ln1_w = nxt(); L.ln1_b = nxt(); L.ln2_w = nxt(); L.ln2_b = nxt();
    if (cross) { rdA(L.ca); L.ln3_w = nxt(); L.ln3_b = nxt(); }
  };
  for (int i = 0; i < 2; ++i) rdL(enc[i], false);
  for (int i = 0; i < 2; ++i) rdL(sdec[i], true);
  for (int i = 0; i < 2; ++i) rdL(dcr[i], true);
  const float* init_tok = nxt();

  // ---- workspace bump allocation ----
  char* p = (char*)d_ws;
  auto cvt = [&](const float* src, u16* dst, long n) {
    e2t_cvt<<<dim3(1024), 256, 0, stream>>>(src, dst, n);
  };
  auto mkW = [&](const LayerP& P, bool cross) {
    LayerW W{};
    W.sa.wqkv = (u16*)bump(p, (size_t)3*D*D*2); cvt(P.sa.in_w,  W.sa.wqkv, 3*D*D);
    W.sa.wo   = (u16*)bump(p, (size_t)D*D*2);   cvt(P.sa.out_w, W.sa.wo,   D*D);
    W.w1      = (u16*)bump(p, (size_t)DFF*D*2); cvt(P.lin1_w,   W.w1,      (long)DFF*D);
    W.w2      = (u16*)bump(p, (size_t)D*DFF*2); cvt(P.lin2_w,   W.w2,      (long)D*DFF);
    if (cross) {
      W.ca.wqkv = (u16*)bump(p, (size_t)3*D*D*2); cvt(P.ca.in_w,  W.ca.wqkv, 3*D*D);
      W.ca.wo   = (u16*)bump(p, (size_t)D*D*2);   cvt(P.ca.out_w, W.ca.wo,   D*D);
    }
    return W;
  };
  LayerW Wenc[2]  = { mkW(enc[0], false),  mkW(enc[1], false)  };
  LayerW Wsdec[2] = { mkW(sdec[0], true),  mkW(sdec[1], true)  };
  LayerW Wdcr[2]  = { mkW(dcr[0], true),   mkW(dcr[1], true)   };

  const long NTOK = (long)BATCH * SEQ;               // 32768 encoder tokens
  float* exf  = (float*)bump(p, NTOK * D * 4);
  u16*   exb  = (u16*)  bump(p, NTOK * D * 2);
  float* x2f  = (float*)bump(p, NTOK * D * 4);
  u16*   x2b  = (u16*)  bump(p, NTOK * D * 2);
  float* tmpf = (float*)bump(p, NTOK * D * 4);
  u16*   eq   = (u16*)  bump(p, NTOK * D * 2);
  u16*   ek   = (u16*)  bump(p, NTOK * D * 2);
  u16*   evT  = (u16*)  bump(p, NTOK * D * 2);
  u16*   eattn= (u16*)  bump(p, NTOK * D * 2);
  float* escore=(float*)bump(p, (long)BATCH*NH*SEQ*SEQ * 4);
  u16*   eprob =(u16*)  bump(p, (long)BATCH*NH*SEQ*SEQ * 2);
  u16*   effn  =(u16*)  bump(p, NTOK * DFF * 2);
  u16 *kmem[4], *vmem[4];
  for (int i = 0; i < 4; ++i) {
    kmem[i] = (u16*)bump(p, NTOK * D * 2);
    vmem[i] = (u16*)bump(p, NTOK * D * 2);
  }
  const long TOKN = (long)BATCH * DCAP * D;
  float* tokf = (float*)bump(p, TOKN * 4);
  u16*   tokb = (u16*)  bump(p, TOKN * 2);
  float* laf  = (float*)bump(p, TOKN * 4);
  u16*   lab  = (u16*)  bump(p, TOKN * 2);
  float* lbf  = (float*)bump(p, TOKN * 4);
  u16*   lbb  = (u16*)  bump(p, TOKN * 2);
  DecScratch s{};
  s.dq    = (u16*)  bump(p, TOKN * 2);
  s.dk    = (u16*)  bump(p, TOKN * 2);
  s.dvT   = (u16*)  bump(p, TOKN * 2);
  s.dattn = (u16*)  bump(p, TOKN * 2);
  s.dffn  = (u16*)  bump(p, (long)BATCH*DCAP*DFF * 2);
  s.dprob = (u16*)  bump(p, (long)BATCH*NH*DCAP*DCAP * 2);
  s.cprob = (u16*)  bump(p, (long)BATCH*NH*DCAP*SEQ * 2);
  s.dscore= (float*)bump(p, (long)BATCH*NH*DCAP*DCAP * 4);
  s.cscore= (float*)bump(p, (long)BATCH*NH*DCAP*SEQ * 4);
  s.dmha  = (float*)bump(p, TOKN * 4);
  s.x2f   = (float*)bump(p, TOKN * 4);  s.x2b = (u16*)bump(p, TOKN * 2);
  s.x3f   = (float*)bump(p, TOKN * 4);  s.x3b = (u16*)bump(p, TOKN * 2);

  // ------------------------------- encoder ---------------------------------
  e2t_copy_cvt<<<dim3(4096), 256, 0, stream>>>(source, exf, exb, NTOK * D);
  for (int li = 0; li < 2; ++li) {
    const LayerP& P = enc[li]; const LayerW& W = Wenc[li];
    // QKV projections over all tokens (flat GEMMs)
    launch_gemm<true,false,true,false>(stream, exb, W.sa.wqkv,         P.sa.in_b,       eq,
        (int)NTOK, D, D, D, D, D, 0,0,0,0,0,0, 1, 1, 1.f);
    launch_gemm<true,false,true,false>(stream, exb, W.sa.wqkv + D*D,   P.sa.in_b + D,   ek,
        (int)NTOK, D, D, D, D, D, 0,0,0,0,0,0, 1, 1, 1.f);
    launch_gemm<true,false,true,true >(stream, exb, W.sa.wqkv + 2*D*D, P.sa.in_b + 2*D, evT,
        (int)NTOK, D, D, D, D, (int)NTOK, 0,0,0,0,0,0, 1, 1, 1.f);
    // scores = QK^T / sqrt(hd), per (b,h)
    launch_gemm<false,false,false,false>(stream, eq, ek, nullptr, escore,
        SEQ, SEQ, HDIM, D, D, SEQ, (long)SEQ*D,(long)HDIM, (long)SEQ*D,(long)HDIM,
        (long)NH*SEQ*SEQ,(long)SEQ*SEQ, NH, BATCH*NH, RSQRT_HD);
    e2t_softmax<<<dim3((SEQ+7)/8,1,BATCH*NH),256,0,stream>>>(escore, eprob,
        SEQ, SEQ, SEQ, (long)SEQ*SEQ, (long)SEQ*SEQ, 0);
    launch_gemm<false,false,true,false>(stream, eprob, evT, nullptr, eattn,
        SEQ, HDIM, SEQ, SEQ, (int)NTOK, D,
        (long)NH*SEQ*SEQ,(long)SEQ*SEQ, (long)SEQ,(long)HDIM*NTOK,
        (long)SEQ*D,(long)HDIM, NH, BATCH*NH, 1.f);
    launch_gemm<true,false,false,false>(stream, eattn, W.sa.wo, P.sa.out_b, tmpf,
        (int)NTOK, D, D, D, D, D, 0,0,0,0,0,0, 1, 1, 1.f);
    e2t_add_ln<<<dim3(((int)NTOK+7)/8),256,0,stream>>>(exf, tmpf, P.ln1_w, P.ln1_b,
        x2f, x2b, (int)NTOK, 0, 0, 0, (int)NTOK);
    // FFN
    launch_gemm<true,true,true,false>(stream, x2b, W.w1, P.lin1_b, effn,
        (int)NTOK, DFF, D, D, D, DFF, 0,0,0,0,0,0, 1, 1, 1.f);
    launch_gemm<true,false,false,false>(stream, effn, W.w2, P.lin2_b, tmpf,
        (int)NTOK, D, DFF, DFF, DFF, D, 0,0,0,0,0,0, 1, 1, 1.f);
    e2t_add_ln<<<dim3(((int)NTOK+7)/8),256,0,stream>>>(x2f, tmpf, P.ln2_w, P.ln2_b,
        exf, exb, (int)NTOK, 0, 0, 0, (int)NTOK);
  }
  // exb now holds mem (bf16). Cache cross-attn K/V of mem per layer (loop-
  // invariant in the reference — computed once here).
  const LayerP* xlayers[4] = { &sdec[0], &sdec[1], &dcr[0], &dcr[1] };
  const LayerW* xw[4]      = { &Wsdec[0], &Wsdec[1], &Wdcr[0], &Wdcr[1] };
  for (int i = 0; i < 4; ++i) {
    launch_gemm<true,false,true,false>(stream, exb, xw[i]->ca.wqkv + D*D,
        xlayers[i]->ca.in_b + D, kmem[i],
        (int)NTOK, D, D, D, D, D, 0,0,0,0,0,0, 1, 1, 1.f);
    launch_gemm<true,false,true,true >(stream, exb, xw[i]->ca.wqkv + 2*D*D,
        xlayers[i]->ca.in_b + 2*D, vmem[i],
        (int)NTOK, D, D, D, D, (int)NTOK, 0,0,0,0,0,0, 1, 1, 1.f);
  }

  // ------------------------------- decoders --------------------------------
  e2t_tok_init<<<dim3(32), 256, 0, stream>>>(init_tok, tokf, tokb);
  // spatial decoder (T=1, no mask); result replaces tok[:,0,:]
  dec_layer(stream, sdec[0], Wsdec[0], kmem[0], vmem[0], 1, 0, tokf, tokb, laf, lab, s);
  dec_layer(stream, sdec[1], Wsdec[1], kmem[1], vmem[1], 1, 0, laf, lab, lbf, lbb, s);
  e2t_append<<<dim3(32), 256, 0, stream>>>(lbf, 0, 0, tokf, tokb);
  // autoregressive decode (causal self-attn), append last token each step
  for (int i = 0; i < NB; ++i) {
    const int T = i + 1;
    dec_layer(stream, dcr[0], Wdcr[0], kmem[2], vmem[2], T, 1, tokf, tokb, laf, lab, s);
    dec_layer(stream, dcr[1], Wdcr[1], kmem[3], vmem[3], T, 1, laf, lab, lbf, lbb, s);
    e2t_append<<<dim3(32), 256, 0, stream>>>(lbf, T - 1, i + 1, tokf, tokb);
  }
  e2t_emit<<<dim3((BATCH*(NB+1)*D + 255)/256), 256, 0, stream>>>(tokf, (float*)d_out);
}
